// PCAProjector_9826885173794
// MI455X (gfx1250) — matrix operand order repaired and checked by reading, not compiled, not run on hardware
//
#include <hip/hip_runtime.h>
#include <hip/hip_bf16.h>

// ---------------------------------------------------------------------------
// PCAProjector for MI455X (gfx1250, wave32, WMMA).
// All large GEMMs: v_wmma_f32_16x16x32_bf16, bf16 operands, fp32 accumulate,
// with register ping-pong double buffering so global loads overlap WMMAs.
// cov = H^T H is expressed as the same GEMM (A = Bt = transposed head slab).
// eigh: parallel cyclic Jacobi in LDS, one 64x64 matrix per workgroup.
// ---------------------------------------------------------------------------

typedef __attribute__((ext_vector_type(16))) __bf16 v16bf;
typedef __attribute__((ext_vector_type(8)))  __bf16 v8bf;
typedef __attribute__((ext_vector_type(8)))  float  v8f;

// ---------------- fragment loaders (layouts per CDNA5 ISA 7.12.2) ----------
// A fragment 16x32 bf16: lane l: row m = l&15, kg = l>>4.
//   elems 0..7  = A[m][k0 + kg*8 + 0..7]
//   elems 8..15 = A[m][k0 + 16 + kg*8 + 0..7]
__device__ __forceinline__ v16bf load_a_frag(const __bf16* A, int lda, int m,
                                             int k0, int kg) {
  const __bf16* p0 = A + (size_t)m * lda + k0 + kg * 8;
  v8bf lo = *(const v8bf*)p0;
  v8bf hi = *(const v8bf*)(p0 + 16);
  v16bf r;
#pragma unroll
  for (int e = 0; e < 8; ++e) { r[e] = lo[e]; r[8 + e] = hi[e]; }
  return r;
}

// B fragment 32x16 bf16 from Bt (N,K) row-major: lane l: col n, kg = l>>4.
//   elems 0..7  = Bt[n][k0 + kg*8 + 0..7], elems 8..15 = Bt[n][k0 + 16 + kg*8 + 0..7]
__device__ __forceinline__ v16bf load_b_frag(const __bf16* Bt, int ldb, int n,
                                             int k0, int kg) {
  const __bf16* p = Bt + (size_t)n * ldb + k0 + kg * 8;
  v8bf lo = *(const v8bf*)p;
  v8bf hi = *(const v8bf*)(p + 16);
  v16bf r;
#pragma unroll
  for (int e = 0; e < 8; ++e) { r[e] = lo[e]; r[8 + e] = hi[e]; }
  return r;
}

// ---------------- WMMA GEMM: C = act(A @ Bt^T + bias) ----------------------
// A: (M,K) bf16 row-major. Bt: (N,K) bf16 row-major. This model guarantees
// M%64==0, N%64==0, K%64==0. Block = 256 thr = 8 waves, wave tile 16x64,
// block tile 64(M) x 128(N). Batched over blockIdx.z.
// C offset = (z/inner)*c_outer + (z%inner)*c_inner + row*c_rs + col.
// flags: 1=relu, 2=add bias, 4=store bf16 (else fp32).
__global__ __launch_bounds__(256) void gemm_wmma_kernel(
    const __bf16* __restrict__ A, const __bf16* __restrict__ Bt,
    const float* __restrict__ bias, void* __restrict__ C,
    int M, int N, int K,
    long long a_bs, long long b_bs,
    int c_rs, long long c_outer, long long c_inner, int inner_cnt, int flags) {
  const int z = blockIdx.z;
  const __bf16* Ab = A + (size_t)z * a_bs;
  const __bf16* Bb = Bt + (size_t)z * b_bs;
  const size_t cbase = (size_t)(z / inner_cnt) * c_outer +
                       (size_t)(z % inner_cnt) * c_inner;

  const int w  = threadIdx.x >> 5;
  const int l  = threadIdx.x & 31;
  const int wm = w & 3;        // 4 M-strips
  const int wn = w >> 2;       // 2 N-groups
  const int m0 = blockIdx.x * 64 + wm * 16;
  const int n0 = blockIdx.y * 128 + wn * 64;
  if (m0 >= M || n0 >= N) return;   // wave-uniform -> EXEC stays all-ones

  const int kg = l >> 4;
  const int ln = l & 15;
  const int m  = m0 + ln;

  v8f acc[4] = {};

  // Register ping-pong double buffering (K is always a multiple of 64):
  // issue the next fragment-set's loads before consuming the current set's
  // WMMAs so the scheduler can hide global latency with partial loadcnt waits.
  v16bf a0 = load_a_frag(Ab, K, m, 0, kg);
  v16bf b0[4];
#pragma unroll
  for (int j = 0; j < 4; ++j) b0[j] = load_b_frag(Bb, K, n0 + j * 16 + ln, 0, kg);

  for (int k0 = 0; k0 < K; k0 += 64) {
    __builtin_prefetch(Ab + (size_t)m * K + k0 + 64, 0, 1);  // global_prefetch_b8
    // load set 1 (k0+32) while set 0 computes
    v16bf a1 = load_a_frag(Ab, K, m, k0 + 32, kg);
    v16bf b1[4];
#pragma unroll
    for (int j = 0; j < 4; ++j)
      b1[j] = load_b_frag(Bb, K, n0 + j * 16 + ln, k0 + 32, kg);
#pragma unroll
    for (int j = 0; j < 4; ++j)
      acc[j] = __builtin_amdgcn_wmma_f32_16x16x32_bf16(
          false, a0, false, b0[j], (short)0, acc[j], false, false);
    // refill set 0 (k0+64) while set 1 computes
    if (k0 + 64 < K) {
      a0 = load_a_frag(Ab, K, m, k0 + 64, kg);
#pragma unroll
      for (int j = 0; j < 4; ++j)
        b0[j] = load_b_frag(Bb, K, n0 + j * 16 + ln, k0 + 64, kg);
    }
#pragma unroll
    for (int j = 0; j < 4; ++j)
      acc[j] = __builtin_amdgcn_wmma_f32_16x16x32_bf16(
          false, a1, false, b1[j], (short)0, acc[j], false, false);
  }

  const bool do_relu = flags & 1, do_bias = flags & 2, out_bf = flags & 4;
#pragma unroll
  for (int j = 0; j < 4; ++j) {
    const int col = n0 + j * 16 + ln;
    const float bb = do_bias ? bias[col] : 0.0f;
#pragma unroll
    for (int r = 0; r < 8; ++r) {
      const int row = m0 + kg * 8 + r;   // C layout: VGPR r -> M = r + 8*kg
      float v = acc[j][r] + bb;
      if (do_relu) v = fmaxf(v, 0.0f);
      const size_t off = cbase + (size_t)row * c_rs + col;
      if (out_bf) ((__bf16*)C)[off] = (__bf16)v;
      else        ((float*)C)[off]  = v;
    }
  }
}

// ---------------- per-head transpose: (B,S,512) -> (B*8, 64, S) ------------
__global__ __launch_bounds__(256) void head_transpose_kernel(
    const __bf16* __restrict__ h, __bf16* __restrict__ ht, int S) {
  __shared__ __bf16 tile[64][65];
  const int z = blockIdx.y;            // b*8 + head
  const int b = z >> 3, head = z & 7;
  const int s0 = blockIdx.x * 64;
  const __bf16* src = h + ((size_t)b * S) * 512 + head * 64;
  const int tx = threadIdx.x & 63, ty = threadIdx.x >> 6;
#pragma unroll
  for (int it = 0; it < 16; ++it) {
    const int row = it * 4 + ty;
    tile[row][tx] = src[(size_t)(s0 + row) * 512 + tx];
  }
  __syncthreads();
  __bf16* dst = ht + (size_t)z * 64 * S;
#pragma unroll
  for (int it = 0; it < 16; ++it) {
    const int row = it * 4 + ty;       // channel index i
    dst[(size_t)row * S + s0 + tx] = tile[tx][row];
  }
}

// ---------------- symmetric eigh via parallel cyclic Jacobi ----------------
// One workgroup (64 thr = 2 waves) per 64x64 matrix. Round-robin ordering:
// 63 rounds x 32 disjoint pairs; each pair served by 2 threads on split
// k-ranges (no write/read races). Output: rows = eigenvectors sorted by
// ascending eigenvalue (matches eigh followed by swapaxes in the reference).
__global__ __launch_bounds__(64) void jacobi_eigh_kernel(
    const float* __restrict__ cov, __bf16* __restrict__ vrow) {
  __shared__ float Am[64][65];
  __shared__ float Vm[64][65];
  __shared__ float cs[32], sn[32];
  __shared__ int   perm[64];
  const int z = blockIdx.x, tid = threadIdx.x;
  const float* src = cov + (size_t)z * 4096;
  for (int i = tid; i < 4096; i += 64) {
    const int r = i >> 6, c = i & 63;
    Am[r][c] = src[i];
    Vm[r][c] = (r == c) ? 1.0f : 0.0f;
  }
  __syncthreads();

  const int pi = tid >> 1;      // pair index 0..31
  const int half = tid & 1;
  for (int sweep = 0; sweep < 12; ++sweep) {
    for (int t = 0; t < 63; ++t) {
      const int ia = pi, ib = 63 - pi;
      const int a = (ia == 0) ? 0 : 1 + ((ia - 1 + t) % 63);
      const int b = 1 + ((ib - 1 + t) % 63);
      const int p = a < b ? a : b;
      const int q = a < b ? b : a;
      if (half == 0) {
        const float app = Am[p][p], aqq = Am[q][q], apq = Am[p][q];
        float c = 1.0f, s = 0.0f;
        if (fabsf(apq) > 1e-12f) {
          const float th = (aqq - app) / (2.0f * apq);
          float tt = 1.0f / (fabsf(th) + sqrtf(th * th + 1.0f));
          if (th < 0.0f) tt = -tt;
          c = rsqrtf(tt * tt + 1.0f);
          s = tt * c;
        }
        cs[pi] = c; sn[pi] = s;
      }
      __syncthreads();
      const float c = cs[pi], s = sn[pi];
      const int k0 = half * 32;
      for (int k = k0; k < k0 + 32; ++k) {       // row rotation (J^T A)
        const float rp = Am[p][k], rq = Am[q][k];
        Am[p][k] = c * rp - s * rq;
        Am[q][k] = s * rp + c * rq;
      }
      __syncthreads();
      for (int k = k0; k < k0 + 32; ++k) {       // col rotation (.. J), V = V J
        const float cp = Am[k][p], cq = Am[k][q];
        Am[k][p] = c * cp - s * cq;
        Am[k][q] = s * cp + c * cq;
        const float vp = Vm[k][p], vq = Vm[k][q];
        Vm[k][p] = c * vp - s * vq;
        Vm[k][q] = s * vp + c * vq;
      }
      __syncthreads();
    }
  }
  if (tid == 0) {                     // sort eigenvalues ascending
    for (int i = 0; i < 64; ++i) perm[i] = i;
    for (int i = 1; i < 64; ++i) {
      const int key = perm[i];
      const float kv = Am[key][key];
      int j = i - 1;
      while (j >= 0 && Am[perm[j]][perm[j]] > kv) { perm[j + 1] = perm[j]; --j; }
      perm[j + 1] = key;
    }
  }
  __syncthreads();
  __bf16* dst = vrow + (size_t)z * 4096;
  for (int i = tid; i < 4096; i += 64) {
    const int r = i >> 6, c = i & 63;
    dst[i] = (__bf16)Vm[c][perm[r]];  // row r = r-th eigenvector
  }
}

// ---------------- small helpers --------------------------------------------
__global__ void f32_to_bf16_kernel(const float* __restrict__ x,
                                   __bf16* __restrict__ y, long long n) {
  const long long i = (long long)blockIdx.x * blockDim.x + threadIdx.x;
  if (i < n) y[i] = (__bf16)x[i];
}

// W (K,N) f32 row-major -> Wt (N,K) bf16 row-major
__global__ void wt_transpose_kernel(const float* __restrict__ W,
                                    __bf16* __restrict__ Wt, int Kd, int Nd) {
  const int i = blockIdx.x * blockDim.x + threadIdx.x;
  if (i < Kd * Nd) {
    const int n = i / Kd, k = i % Kd;
    Wt[i] = (__bf16)W[(size_t)k * Nd + n];
  }
}

// p[r] = relu(y1[r,:] . Wpoint + bpoint); one wave32 per row, coalesced
// 32B-per-lane chunks + shfl_xor reduction.
__global__ __launch_bounds__(256) void point_kernel(
    const __bf16* __restrict__ y, const float* __restrict__ w,
    const float* __restrict__ b, float* __restrict__ p, int rows) {
  const int wid  = (blockIdx.x * blockDim.x + threadIdx.x) >> 5;
  const int lane = threadIdx.x & 31;
  if (wid >= rows) return;
  const __bf16* yr = y + (size_t)wid * 512 + lane * 16;
  v8bf u0 = *(const v8bf*)yr;
  v8bf u1 = *(const v8bf*)(yr + 8);
  const float* wr = w + lane * 16;
  float acc = 0.0f;
#pragma unroll
  for (int e = 0; e < 8; ++e)
    acc += (float)u0[e] * wr[e] + (float)u1[e] * wr[8 + e];
#pragma unroll
  for (int off = 16; off > 0; off >>= 1) acc += __shfl_xor(acc, off, 32);
  if (lane == 0) p[wid] = fmaxf(acc + b[0], 0.0f);
}

// out[b,e] = relu(p[b,:] . Wback[:,e] + bback[e])   (16 x 128 outputs)
__global__ void back_kernel(const float* __restrict__ p,
                            const float* __restrict__ Wback,
                            const float* __restrict__ bback,
                            float* __restrict__ out) {
  const int i = blockIdx.x * blockDim.x + threadIdx.x;
  if (i >= 16 * 128) return;
  const int b = i >> 7, e = i & 127;
  const float* pr = p + b * 512;
  float acc = 0.0f;
  for (int k = 0; k < 512; ++k) acc += pr[k] * Wback[k * 128 + e];
  out[i] = fmaxf(acc + bback[e], 0.0f);
}

// ---------------------------------------------------------------------------
extern "C" void kernel_launch(void* const* d_in, const int* in_sizes, int n_in,
                              void* d_out, int out_size, void* d_ws,
                              size_t ws_size, hipStream_t stream) {
  const float* x      = (const float*)d_in[0];
  const float* Wd2_0  = (const float*)d_in[1];
  const float* bd2_0  = (const float*)d_in[2];
  const float* Wup_0  = (const float*)d_in[3];
  const float* bup_0  = (const float*)d_in[4];
  const float* Wdff_0 = (const float*)d_in[5];
  const float* bdff_0 = (const float*)d_in[6];
  const float* Wff_0  = (const float*)d_in[7];
  const float* bff_0  = (const float*)d_in[8];
  const float* Wd2_1  = (const float*)d_in[9];
  const float* bd2_1  = (const float*)d_in[10];
  const float* Wup_1  = (const float*)d_in[11];
  const float* bup_1  = (const float*)d_in[12];
  const float* Wdff_1 = (const float*)d_in[13];
  const float* bdff_1 = (const float*)d_in[14];
  const float* Wff_1  = (const float*)d_in[15];
  const float* bff_1  = (const float*)d_in[16];
  const float* Wpoint = (const float*)d_in[17];
  const float* bpoint = (const float*)d_in[18];
  const float* Wback  = (const float*)d_in[19];
  const float* bback  = (const float*)d_in[20];

  const int B = 16, S = 4096, M0 = B * S /*65536*/, M1 = B * 512 /*8192*/;
  char* ws = (char*)d_ws;
  const size_t MB = 1024 * 1024;

  // ---- workspace layout (bytes), with reuse; peak ~192 MB ----
  __bf16* wd2_0t  = (__bf16*)(ws + 0);
  __bf16* wup_0t  = (__bf16*)(ws + 65536);
  __bf16* wdff_0t = (__bf16*)(ws + 327680);
  __bf16* wff_0t  = (__bf16*)(ws + 393216);
  __bf16* wd2_1t  = (__bf16*)(ws + 917504);
  __bf16* wup_1t  = (__bf16*)(ws + 1179648);
  __bf16* wdff_1t = (__bf16*)(ws + 1441792);
  __bf16* wff_1t  = (__bf16*)(ws + 1507328);
  __bf16* xb   = (__bf16*)(ws + 4 * MB);     // 16.8 MB
  __bf16* h1   = (__bf16*)(ws + 21 * MB);    // 33.6 MB (dead after h2)
  __bf16* h2   = (__bf16*)(ws + 56 * MB);    // 67.1 MB (dead after transpose)
  __bf16* ht   = (__bf16*)(ws + 124 * MB);   // 67.1 MB (dead after cov)
  float*  cov  = (float*) (ws + 21 * MB);    // 2 MB, reuses dead h1 region
  __bf16* vrow = (__bf16*)(ws + 24 * MB);    // 1 MB,  reuses dead h1 region
  __bf16* mh   = (__bf16*)(ws + 56 * MB);    // reuses dead h2 region
  __bf16* y0   = (__bf16*)(ws + 65 * MB);
  __bf16* h1b  = (__bf16*)(ws + 74 * MB);
  __bf16* h2b  = (__bf16*)(ws + 79 * MB);
  __bf16* ht1  = (__bf16*)(ws + 88 * MB);
  float*  cov1 = (float*) (ws + 97 * MB);
  __bf16* vrow1= (__bf16*)(ws + 100 * MB);
  __bf16* mh1  = (__bf16*)(ws + 102 * MB);
  __bf16* y1   = (__bf16*)(ws + 111 * MB);
  float*  pbuf = (float*) (ws + 120 * MB);

  auto gemm = [&](const __bf16* A, const __bf16* Bt, const float* bias,
                  void* C, int M, int N, int K, int batch,
                  long long a_bs, long long b_bs, int c_rs,
                  long long c_outer, long long c_inner, int inner_cnt,
                  int flags) {
    dim3 g((M + 63) / 64, (N + 127) / 128, batch);
    gemm_wmma_kernel<<<g, 256, 0, stream>>>(A, Bt, bias, C, M, N, K, a_bs,
                                            b_bs, c_rs, c_outer, c_inner,
                                            inner_cnt, flags);
  };

  // ---- stage 0: conversions ----
  {
    long long n = (long long)M0 * 128;
    f32_to_bf16_kernel<<<(unsigned)((n + 255) / 256), 256, 0, stream>>>(x, xb, n);
  }
  wt_transpose_kernel<<<(128 * 256 + 255) / 256, 256, 0, stream>>>(Wd2_0, wd2_0t, 128, 256);
  wt_transpose_kernel<<<(256 * 512 + 255) / 256, 256, 0, stream>>>(Wup_0, wup_0t, 256, 512);
  wt_transpose_kernel<<<(64 * 512 + 255) / 256, 256, 0, stream>>>(Wdff_0, wdff_0t, 64, 512);
  wt_transpose_kernel<<<(512 * 512 + 255) / 256, 256, 0, stream>>>(Wff_0, wff_0t, 512, 512);
  wt_transpose_kernel<<<(512 * 256 + 255) / 256, 256, 0, stream>>>(Wd2_1, wd2_1t, 512, 256);
  wt_transpose_kernel<<<(256 * 512 + 255) / 256, 256, 0, stream>>>(Wup_1, wup_1t, 256, 512);
  wt_transpose_kernel<<<(64 * 512 + 255) / 256, 256, 0, stream>>>(Wdff_1, wdff_1t, 64, 512);
  wt_transpose_kernel<<<(512 * 512 + 255) / 256, 256, 0, stream>>>(Wff_1, wff_1t, 512, 512);

  // ---- layer 0 mhpca ----
  gemm(xb, wd2_0t, bd2_0, h1, M0, 256, 128, 1, 0, 0, 256, 0, 0, 1, 7);
  gemm(h1, wup_0t, bup_0, h2, M0, 512, 256, 1, 0, 0, 512, 0, 0, 1, 7);
  head_transpose_kernel<<<dim3(S / 64, B * 8), 256, 0, stream>>>(h2, ht, S);
  // cov = Ht @ Ht^T  (batched 128x; A = Bt = head slab, fp32 out, no act)
  gemm(ht, ht, nullptr, cov, 64, 64, S, B * 8,
       (long long)64 * S, (long long)64 * S, 64, 4096, 0, 1, 0);
  jacobi_eigh_kernel<<<B * 8, 64, 0, stream>>>(cov, vrow);
  // out(b,head): row i of relu(vrow @ Wdff) -> mh[b][i*8+head][:]
  gemm(vrow, wdff_0t, bdff_0, mh, 64, 512, 64, B * 8,
       4096, 0, 8 * 512, (long long)512 * 512, 512, 8, 7);
  gemm(mh, wff_0t, bff_0, y0, M1, 512, 512, 1, 0, 0, 512, 0, 0, 1, 7);

  // ---- layer 1 mhpca (N = 512) ----
  gemm(y0, wd2_1t, bd2_1, h1b, M1, 256, 512, 1, 0, 0, 256, 0, 0, 1, 7);
  gemm(h1b, wup_1t, bup_1, h2b, M1, 512, 256, 1, 0, 0, 512, 0, 0, 1, 7);
  head_transpose_kernel<<<dim3(512 / 64, B * 8), 256, 0, stream>>>(h2b, ht1, 512);
  gemm(ht1, ht1, nullptr, cov1, 64, 64, 512, B * 8,
       (long long)64 * 512, (long long)64 * 512, 64, 4096, 0, 1, 0);
  jacobi_eigh_kernel<<<B * 8, 64, 0, stream>>>(cov1, vrow1);
  gemm(vrow1, wdff_1t, bdff_1, mh1, 64, 512, 64, B * 8,
       4096, 0, 8 * 512, (long long)512 * 512, 512, 8, 7);
  gemm(mh1, wff_1t, bff_1, y1, M1, 512, 512, 1, 0, 0, 512, 0, 0, 1, 7);

  // ---- head ----
  point_kernel<<<(M1 * 32 + 255) / 256, 256, 0, stream>>>(y1, Wpoint, bpoint,
                                                          pbuf, M1);
  back_kernel<<<(2048 + 255) / 256, 256, 0, stream>>>(pbuf, Wback, bback,
                                                      (float*)d_out);
}
